// KGEncoder_10488310137069
// MI455X (gfx1250) — matrix operand order repaired and checked by reading, not compiled, not run on hardware
//
#include <hip/hip_runtime.h>
#include <hip/hip_bf16.h>

#define NN 50000
#define NNP 50048              // NN rounded up to 128 (padded rows for xl/xr)
#define EE 600000
#define EEP 600064             // EE rounded up to 128 (padded scores)
#define DD 128
#define GG 512
#define QQ 3
#define KK 512
#define FDD 512
#define NEG_SLOPE 0.2f
#define LN_EPS 1e-5f

typedef _Float16 half_t;
typedef __attribute__((ext_vector_type(4)))  _Float16 v4h;
typedef __attribute__((ext_vector_type(8)))  _Float16 v8h;
typedef __attribute__((ext_vector_type(16))) _Float16 v16h;
typedef __attribute__((ext_vector_type(8)))  float    v8f;

static __device__ __forceinline__ v8f wmma_f16(v16h a, v16h b, v8f c) {
  return __builtin_amdgcn_wmma_f32_16x16x32_f16(false, a, false, b, (short)0, c, false, false);
}

// A fragment: 16 rows x 32 K from an LDS f16 tile with row stride ldm.
static __device__ __forceinline__ v16h load_a_lds(const half_t* s, int ldm, int rowbase, int k0) {
  int lane = threadIdx.x & 31;
  int row  = rowbase + (lane & 15);
  int kh   = (lane >> 4) << 3;                 // 0 or 8
  const half_t* p = s + row * ldm + k0;
  v8h lo = *(const v8h*)(p + kh);
  v8h hi = *(const v8h*)(p + 16 + kh);
  return __builtin_shufflevector(lo, hi, 0, 1, 2, 3, 4, 5, 6, 7,
                                 8, 9, 10, 11, 12, 13, 14, 15);
}

// B fragment from fragment-major packed weights: one contiguous 32B per-lane load.
static __device__ __forceinline__ v16h load_b_pk(const half_t* pk, int nCT, int kc, int ct) {
  int lane = threadIdx.x & 31;
  return *(const v16h*)(pk + (((size_t)(kc * nCT + ct) * 32 + lane) << 4));
}

static __device__ __forceinline__ void atomicMaxF(float* a, float v) {
  if (v >= 0.0f) atomicMax((int*)a, __float_as_int(v));
  else           atomicMin((unsigned int*)a, __float_as_uint(v));
}

// pack row-major f32 weight [K][N] into WMMA-B fragment-major f16
static __device__ void pack_frag(const float* __restrict__ src, half_t* __restrict__ dst,
                                 int K, int N, int t, int stride) {
  int nCT = N >> 4;
  for (int i = t; i < K * N; i += stride) {
    int e = i & 15, lane = (i >> 4) & 31, tt = i >> 9;
    int ct = tt % nCT, kc = tt / nCT;
    int col = ct * 16 + (lane & 15);
    int k = kc * 32 + ((e >> 3) << 4) + ((lane >> 4) << 3) + (e & 7);
    dst[i] = (half_t)src[(size_t)k * N + col];
  }
}

// pack transposed source: B[k][col] = src[col*K + k]  (codebooks)
static __device__ void pack_frag_T(const float* __restrict__ src, half_t* __restrict__ dst,
                                   int K, int N, int t, int stride) {
  int nCT = N >> 4;
  for (int i = t; i < K * N; i += stride) {
    int e = i & 15, lane = (i >> 4) & 31, tt = i >> 9;
    int ct = tt % nCT, kc = tt / nCT;
    int col = ct * 16 + (lane & 15);
    int k = kc * 32 + ((e >> 3) << 4) + ((lane >> 4) << 3) + (e & 7);
    dst[i] = (half_t)src[(size_t)col * K + k];
  }
}

// ---------------------------------------------------------------- init
__global__ void k_init(float* __restrict__ smax, float* __restrict__ denom,
                       float* __restrict__ outh, float* __restrict__ gsum,
                       float* __restrict__ gcnt, float* __restrict__ colsum,
                       float* __restrict__ lacc) {
  long long i = (long long)blockIdx.x * blockDim.x + threadIdx.x;
  if (i < (long long)NN * DD) outh[i] = 0.0f;
  if (i <= NN) { smax[i] = -3.4e38f; denom[i] = 0.0f; }   // +1 dummy slot
  if (i < GG * DD) gsum[i] = 0.0f;
  if (i < GG) gcnt[i] = 0.0f;
  if (i < DD) colsum[i] = 0.0f;
  if (i < 4) lacc[i] = 0.0f;
}

// ------------------------------------------------- weight prep (fragment-major f16 packs)
__global__ void k_prep(const float* __restrict__ Wl, const float* __restrict__ Wr,
                       const float* __restrict__ We, const float* __restrict__ Wad,
                       const float* __restrict__ Wout, const float* __restrict__ cb,
                       half_t* __restrict__ WlPk, half_t* __restrict__ WrPk,
                       half_t* __restrict__ WePk, half_t* __restrict__ WadPk,
                       half_t* __restrict__ WoutPk, half_t* __restrict__ cbPk,
                       float* __restrict__ cbsq) {
  int stride = gridDim.x * blockDim.x;
  int t = blockIdx.x * blockDim.x + threadIdx.x;
  pack_frag(Wl,  WlPk,  DD, DD,  t, stride);
  pack_frag(Wr,  WrPk,  DD, DD,  t, stride);
  pack_frag(We,  WePk,  DD, DD,  t, stride);
  pack_frag(Wad, WadPk, DD, DD,  t, stride);
  pack_frag(Wout, WoutPk, DD, FDD, t, stride);
  for (int q = 0; q < QQ; ++q)
    pack_frag_T(cb + (size_t)q * KK * DD, cbPk + (size_t)q * DD * KK, DD, KK, t, stride);
  for (int i = t; i < QQ * KK; i += stride) {
    int q = i / KK; int n = i - q * KK;
    const float* row = cb + (size_t)q * KK * DD + (size_t)n * DD;
    float s = 0.0f;
    for (int d = 0; d < DD; ++d) s += row[d] * row[d];
    cbsq[i] = s;
  }
}

// ------------------------------------------------- edge_attr column sum
__global__ void k_colsum(const float* __restrict__ ea, float* __restrict__ colsum) {
  __shared__ float cs[DD];
  int t = threadIdx.x;
  if (t < DD) cs[t] = 0.0f;
  __syncthreads();
  int c4 = t & 31;
  int rofs = t >> 5;
  float4 acc = make_float4(0.f, 0.f, 0.f, 0.f);
  for (int r0 = blockIdx.x * 8; r0 < EE; r0 += gridDim.x * 8) {
    int r = r0 + rofs;
    if (r < EE) {
      float4 v = ((const float4*)ea)[(size_t)r * 32 + c4];
      acc.x += v.x; acc.y += v.y; acc.z += v.z; acc.w += v.w;
    }
  }
  atomicAdd(&cs[c4 * 4 + 0], acc.x);
  atomicAdd(&cs[c4 * 4 + 1], acc.y);
  atomicAdd(&cs[c4 * 4 + 2], acc.z);
  atomicAdd(&cs[c4 * 4 + 3], acc.w);
  __syncthreads();
  if (t < DD) atomicAdd(&colsum[t], cs[t]);
}

// me_emb = mean_attr @ W_edge (fp32 exact)
__global__ void k_meanemb(const float* __restrict__ colsum, const float* __restrict__ We,
                          float* __restrict__ me) {
  int j = threadIdx.x;
  float acc = 0.0f;
  for (int k = 0; k < DD; ++k) acc += (colsum[k] * (1.0f / EE)) * We[k * DD + j];
  me[j] = acc;
}

// ------------------------------------------------- node transforms: xl = x@Wl+bl, xr = x@Wr+br
__global__ void k_nodexf(const float* __restrict__ x, const half_t* __restrict__ WlPk,
                         const half_t* __restrict__ WrPk, const float* __restrict__ bl,
                         const float* __restrict__ br, float* __restrict__ xl,
                         float* __restrict__ xr) {
  __shared__ __align__(32) half_t xs[128 * 128];
  int base = blockIdx.x * 128;
  int tid = threadIdx.x;
  for (int i = tid; i < 128 * 32; i += 256) {
    int r = i >> 5, c4 = i & 31;
    int gr = base + r;
    float4 v = make_float4(0.f, 0.f, 0.f, 0.f);
    if (gr < NN) v = ((const float4*)x)[(size_t)gr * 32 + c4];
    v4h h = {(half_t)v.x, (half_t)v.y, (half_t)v.z, (half_t)v.w};
    *(v4h*)(xs + r * 128 + c4 * 4) = h;
  }
  __syncthreads();
  int w = tid >> 5, lane = tid & 31, rb = w << 4;
  int ro = (lane >> 4) << 3;
  v16h a4[4];
#pragma unroll
  for (int kk = 0; kk < 4; ++kk) a4[kk] = load_a_lds(xs, 128, rb, kk * 32);
  const half_t* Wp[2] = {WlPk, WrPk};
  const float*  bp[2] = {bl, br};
  float*        op[2] = {xl, xr};
#pragma unroll
  for (int m = 0; m < 2; ++m) {
    const half_t* W = Wp[m];
    const float*  bias = bp[m];
    float*        outp = op[m];
    auto store = [&](v8f acc, int ct) {
      int colb = ct * 16 + (lane & 15);
      float bv = bias[colb];                     // hoisted: one load per tile
#pragma unroll
      for (int i = 0; i < 8; ++i)
        outp[(size_t)(base + rb + i + ro) * 128 + colb] = acc[i] + bv;
    };
    v16h bA[4], bB[4];
#pragma unroll
    for (int kk = 0; kk < 4; ++kk) bA[kk] = load_b_pk(W, 8, kk, 0);
#pragma unroll
    for (int cp = 0; cp < 4; ++cp) {
      int ct0 = cp * 2;
#pragma unroll
      for (int kk = 0; kk < 4; ++kk) bB[kk] = load_b_pk(W, 8, kk, ct0 + 1);
      v8f acc = {};
#pragma unroll
      for (int kk = 0; kk < 4; ++kk) acc = wmma_f16(a4[kk], bA[kk], acc);
      store(acc, ct0);
      if (cp < 3) {
#pragma unroll
        for (int kk = 0; kk < 4; ++kk) bA[kk] = load_b_pk(W, 8, kk, ct0 + 2);
      }
      v8f acc2 = {};
#pragma unroll
      for (int kk = 0; kk < 4; ++kk) acc2 = wmma_f16(a4[kk], bB[kk], acc2);
      store(acc2, ct0 + 1);
    }
  }
}

// ------------------------------------------------- fused edge GEMM + attention score + segmax
__global__ void k_edgescore(const float* __restrict__ ea, const int* __restrict__ ei,
                            const half_t* __restrict__ WePk, const float* __restrict__ xl,
                            const float* __restrict__ xr, const float* __restrict__ att,
                            float* __restrict__ scores, float* __restrict__ smax) {
  __shared__ __align__(32) half_t as[128 * 128];
  __shared__ int   si[128], di[128];
  __shared__ float attv[128];
  int base = blockIdx.x * 128;
  int tid = threadIdx.x;
  for (int i = tid; i < 128 * 32; i += 256) {
    int r = i >> 5, c4 = i & 31;
    int ge = base + r;
    float4 v = make_float4(0.f, 0.f, 0.f, 0.f);
    if (ge < EE) v = ((const float4*)ea)[(size_t)ge * 32 + c4];
    v4h h = {(half_t)v.x, (half_t)v.y, (half_t)v.z, (half_t)v.w};
    *(v4h*)(as + r * 128 + c4 * 4) = h;
  }
  if (tid < 128) {
    int ge = base + tid;
    si[tid] = (ge < EE) ? ei[ge] : 0;
    di[tid] = (ge < EE) ? ei[EE + ge] : NN;   // dummy segment for padded edges
    attv[tid] = att[tid];
  }
  if (tid == 0) __builtin_prefetch(ea + (size_t)(base + 128) * 128, 0, 1);
  __syncthreads();
  int w = tid >> 5, lane = tid & 31, rb = w << 4;
  int ro = (lane >> 4) << 3;
  v16h a4[4];
#pragma unroll
  for (int kk = 0; kk < 4; ++kk) a4[kk] = load_a_lds(as, 128, rb, kk * 32);
  float s[8];
#pragma unroll
  for (int i = 0; i < 8; ++i) s[i] = 0.0f;
  auto consume = [&](v8f acc, int ct) {
    int c = ct * 16 + (lane & 15);
    float av = attv[c];
#pragma unroll
    for (int i = 0; i < 8; ++i) {
      int r = rb + i + ro;
      float v = acc[i] + xl[(size_t)si[r] * 128 + c] + xr[(size_t)di[r] * 128 + c];
      float lr = (v > 0.0f) ? v : NEG_SLOPE * v;
      s[i] += lr * av;
    }
  };
  v16h bA[4], bB[4];
#pragma unroll
  for (int kk = 0; kk < 4; ++kk) bA[kk] = load_b_pk(WePk, 8, kk, 0);
#pragma unroll
  for (int cp = 0; cp < 4; ++cp) {
    int ct0 = cp * 2;
#pragma unroll
    for (int kk = 0; kk < 4; ++kk) bB[kk] = load_b_pk(WePk, 8, kk, ct0 + 1);
    v8f acc = {};
#pragma unroll
    for (int kk = 0; kk < 4; ++kk) acc = wmma_f16(a4[kk], bA[kk], acc);
    consume(acc, ct0);
    if (cp < 3) {
#pragma unroll
      for (int kk = 0; kk < 4; ++kk) bA[kk] = load_b_pk(WePk, 8, kk, ct0 + 2);
    }
    v8f acc2 = {};
#pragma unroll
    for (int kk = 0; kk < 4; ++kk) acc2 = wmma_f16(a4[kk], bB[kk], acc2);
    consume(acc2, ct0 + 1);
  }
#pragma unroll
  for (int m = 1; m <= 8; m <<= 1)
#pragma unroll
    for (int i = 0; i < 8; ++i) s[i] += __shfl_xor(s[i], m, 32);
  if ((lane & 15) == 0) {
#pragma unroll
    for (int i = 0; i < 8; ++i) {
      int r = rb + i + ro;
      scores[base + r] = s[i];
      atomicMaxF(&smax[di[r]], s[i]);
    }
  }
}

// self-loop scores (one wave per node)
__global__ void k_selfscore(const float* __restrict__ xl, const float* __restrict__ xr,
                            const float* __restrict__ me, const float* __restrict__ att,
                            float* __restrict__ ssel, float* __restrict__ smax) {
  int wid = (blockIdx.x * blockDim.x + threadIdx.x) >> 5;
  int lane = threadIdx.x & 31;
  if (wid >= NN) return;
  float s = 0.0f;
#pragma unroll
  for (int j = 0; j < 4; ++j) {
    int c = lane + 32 * j;
    float v = xl[(size_t)wid * 128 + c] + xr[(size_t)wid * 128 + c] + me[c];
    float lr = (v > 0.0f) ? v : NEG_SLOPE * v;
    s += lr * att[c];
  }
#pragma unroll
  for (int m = 16; m >= 1; m >>= 1) s += __shfl_xor(s, m, 32);
  if (lane == 0) { ssel[wid] = s; atomicMaxF(&smax[wid], s); }
}

// exp + denom (edges and self-loops)
__global__ void k_pass2(const float* __restrict__ scores, const float* __restrict__ ssel,
                        const int* __restrict__ ei, const float* __restrict__ smax,
                        float* __restrict__ exv, float* __restrict__ exsel,
                        float* __restrict__ denom) {
  long long i = (long long)blockIdx.x * blockDim.x + threadIdx.x;
  if (i < EE) {
    int d = ei[EE + i];
    float e = __expf(scores[i] - smax[d]);
    exv[i] = e;
    atomicAdd(&denom[d], e);
  }
  if (i < NN) {
    float e = __expf(ssel[i] - smax[i]);
    exsel[i] = e;
    atomicAdd(&denom[i], e);
  }
}

// scatter: out[dst] += xl[src] * alpha  (float4 per thread)
__global__ void k_pass3e(const int* __restrict__ ei, const float* __restrict__ exv,
                         const float* __restrict__ denom, const float* __restrict__ xl,
                         float* __restrict__ outh) {
  long long i = (long long)blockIdx.x * blockDim.x + threadIdx.x;
  if (i >= (long long)EE * 32) return;
  int e = (int)(i >> 5), c4 = (int)(i & 31);
  int s = ei[e], d = ei[EE + e];
  float a = exv[e] / (denom[d] + 1e-16f);
  float4 v = ((const float4*)xl)[(size_t)s * 32 + c4];
  float* po = outh + (size_t)d * 128 + c4 * 4;
  atomicAdd(po + 0, v.x * a);
  atomicAdd(po + 1, v.y * a);
  atomicAdd(po + 2, v.z * a);
  atomicAdd(po + 3, v.w * a);
}

__global__ void k_pass3s(const float* __restrict__ exsel, const float* __restrict__ denom,
                         const float* __restrict__ xl, float* __restrict__ outh) {
  long long i = (long long)blockIdx.x * blockDim.x + threadIdx.x;
  if (i >= (long long)NN * 32) return;
  int n = (int)(i >> 5), c4 = (int)(i & 31);
  float a = exsel[n] / (denom[n] + 1e-16f);
  float4 v = ((const float4*)xl)[(size_t)n * 32 + c4];
  float* po = outh + (size_t)n * 128 + c4 * 4;
  atomicAdd(po + 0, v.x * a);
  atomicAdd(po + 1, v.y * a);
  atomicAdd(po + 2, v.z * a);
  atomicAdd(po + 3, v.w * a);
}

// global mean pool (sums + counts)
__global__ void k_pool(const float* __restrict__ outh, const float* __restrict__ cbias,
                       const int* __restrict__ batch, float* __restrict__ gsum,
                       float* __restrict__ gcnt) {
  long long i = (long long)blockIdx.x * blockDim.x + threadIdx.x;
  if (i < (long long)NN * 32) {
    int n = (int)(i >> 5), c4 = (int)(i & 31);
    int b = batch[n];
    float4 v = ((const float4*)outh)[(size_t)n * 32 + c4];
    float4 cb4 = ((const float4*)cbias)[c4];
    float* pg = gsum + (size_t)b * 128 + c4 * 4;
    atomicAdd(pg + 0, v.x + cb4.x);
    atomicAdd(pg + 1, v.y + cb4.y);
    atomicAdd(pg + 2, v.z + cb4.z);
    atomicAdd(pg + 3, v.w + cb4.w);
  }
  if (i < NN) atomicAdd(&gcnt[batch[i]], 1.0f);
}

// adapter GEMM: g = (gsum/cnt) @ adapter_W + b
__global__ void k_adapter(const float* __restrict__ gsum, const float* __restrict__ gcnt,
                          const half_t* __restrict__ WadPk, const float* __restrict__ ab,
                          float* __restrict__ gn) {
  __shared__ __align__(32) half_t ga[128 * 128];
  __shared__ float cnts[128];
  int base = blockIdx.x * 128;
  int tid = threadIdx.x;
  if (tid < 128) cnts[tid] = fmaxf(gcnt[base + tid], 1.0f);
  __syncthreads();
  for (int i = tid; i < 128 * 32; i += 256) {
    int r = i >> 5, c4 = i & 31;
    float4 v = ((const float4*)gsum)[(size_t)(base + r) * 32 + c4];
    float inv = 1.0f / cnts[r];
    v4h h = {(half_t)(v.x * inv), (half_t)(v.y * inv), (half_t)(v.z * inv), (half_t)(v.w * inv)};
    *(v4h*)(ga + r * 128 + c4 * 4) = h;
  }
  __syncthreads();
  int w = tid >> 5, lane = tid & 31, rb = w << 4, ro = (lane >> 4) << 3;
  v16h a4[4];
#pragma unroll
  for (int kk = 0; kk < 4; ++kk) a4[kk] = load_a_lds(ga, 128, rb, kk * 32);
  auto store = [&](v8f acc, int ct) {
    int colb = ct * 16 + (lane & 15);
    float bv = ab[colb];
#pragma unroll
    for (int i = 0; i < 8; ++i)
      gn[(size_t)(base + rb + i + ro) * 128 + colb] = acc[i] + bv;
  };
  v16h bA[4], bB[4];
#pragma unroll
  for (int kk = 0; kk < 4; ++kk) bA[kk] = load_b_pk(WadPk, 8, kk, 0);
#pragma unroll
  for (int cp = 0; cp < 4; ++cp) {
    int ct0 = cp * 2;
#pragma unroll
    for (int kk = 0; kk < 4; ++kk) bB[kk] = load_b_pk(WadPk, 8, kk, ct0 + 1);
    v8f acc = {};
#pragma unroll
    for (int kk = 0; kk < 4; ++kk) acc = wmma_f16(a4[kk], bA[kk], acc);
    store(acc, ct0);
    if (cp < 3) {
#pragma unroll
      for (int kk = 0; kk < 4; ++kk) bA[kk] = load_b_pk(WadPk, 8, kk, ct0 + 2);
    }
    v8f acc2 = {};
#pragma unroll
    for (int kk = 0; kk < 4; ++kk) acc2 = wmma_f16(a4[kk], bB[kk], acc2);
    store(acc2, ct0 + 1);
  }
}

// layernorm over D; also seed residual = g, quant = 0
__global__ void k_adln(float* __restrict__ gn, const float* __restrict__ ng,
                       const float* __restrict__ nb, float* __restrict__ residual,
                       float* __restrict__ quant) {
  __shared__ float red[128];
  int row = blockIdx.x, t = threadIdx.x;
  float x = gn[(size_t)row * 128 + t];
  red[t] = x; __syncthreads();
  for (int s = 64; s > 0; s >>= 1) { if (t < s) red[t] += red[t + s]; __syncthreads(); }
  float m = red[0] * (1.0f / 128.0f); __syncthreads();
  float d = x - m;
  red[t] = d * d; __syncthreads();
  for (int s = 64; s > 0; s >>= 1) { if (t < s) red[t] += red[t + s]; __syncthreads(); }
  float v = red[0] * (1.0f / 128.0f);
  float y = d * rsqrtf(v + LN_EPS) * ng[t] + nb[t];
  size_t o = (size_t)row * 128 + t;
  gn[o] = y; residual[o] = y; quant[o] = 0.0f;
}

// one residual-VQ step (WMMA distance GEMM + argmin + straight-through update)
__global__ void k_vq(int q, float* __restrict__ residual, const half_t* __restrict__ cbPk,
                     const float* __restrict__ cbsq, const float* __restrict__ cbf,
                     float* __restrict__ quant, float* __restrict__ tin,
                     int* __restrict__ idxbuf, float* __restrict__ commit) {
  __shared__ __align__(32) half_t ar[16 * 128];
  __shared__ float arf[16 * 128];
  __shared__ float minv[8][16];
  __shared__ int   mini[8][16];
  __shared__ int   fidx[16];
  __shared__ float cred[256];
  int base = blockIdx.x * 16;
  int tid = threadIdx.x;
  for (int i = tid; i < 16 * 32; i += 256) {
    int r = i >> 5, c4 = i & 31;
    float4 v = ((const float4*)residual)[(size_t)(base + r) * 32 + c4];
    *(float4*)(arf + r * 128 + c4 * 4) = v;
    v4h h = {(half_t)v.x, (half_t)v.y, (half_t)v.z, (half_t)v.w};
    *(v4h*)(ar + r * 128 + c4 * 4) = h;
  }
  __syncthreads();
  int w = tid >> 5, lane = tid & 31, ro = (lane >> 4) << 3;
  v16h a4[4];
#pragma unroll
  for (int kk = 0; kk < 4; ++kk) a4[kk] = load_a_lds(ar, 128, 0, kk * 32);
  const half_t* B = cbPk + (size_t)q * DD * KK;
  const float* sq = cbsq + q * KK;
  float best[8]; int bidx[8];
#pragma unroll
  for (int i = 0; i < 8; ++i) { best[i] = 3.4e38f; bidx[i] = 0; }
  auto consume = [&](v8f acc, int ct) {
    int code = ct * 16 + (lane & 15);
    float csq = sq[code];
#pragma unroll
    for (int i = 0; i < 8; ++i) {
      float d2 = csq - 2.0f * acc[i];
      if (d2 < best[i]) { best[i] = d2; bidx[i] = code; }
    }
  };
  v16h bA[4], bB[4];
#pragma unroll
  for (int kk = 0; kk < 4; ++kk) bA[kk] = load_b_pk(B, 32, kk, w * 4);
#pragma unroll
  for (int cp = 0; cp < 2; ++cp) {
    int ct0 = w * 4 + cp * 2;
#pragma unroll
    for (int kk = 0; kk < 4; ++kk) bB[kk] = load_b_pk(B, 32, kk, ct0 + 1);
    v8f acc = {};
#pragma unroll
    for (int kk = 0; kk < 4; ++kk) acc = wmma_f16(a4[kk], bA[kk], acc);
    consume(acc, ct0);
    if (cp < 1) {
#pragma unroll
      for (int kk = 0; kk < 4; ++kk) bA[kk] = load_b_pk(B, 32, kk, ct0 + 2);
    }
    v8f acc2 = {};
#pragma unroll
    for (int kk = 0; kk < 4; ++kk) acc2 = wmma_f16(a4[kk], bB[kk], acc2);
    consume(acc2, ct0 + 1);
  }
#pragma unroll
  for (int m = 1; m <= 8; m <<= 1)
#pragma unroll
    for (int i = 0; i < 8; ++i) {
      float ov = __shfl_xor(best[i], m, 32);
      int   oi = __shfl_xor(bidx[i], m, 32);
      if (ov < best[i]) { best[i] = ov; bidx[i] = oi; }
    }
  if ((lane & 15) == 0)
#pragma unroll
    for (int i = 0; i < 8; ++i) { minv[w][i + ro] = best[i]; mini[w][i + ro] = bidx[i]; }
  __syncthreads();
  if (tid < 16) {
    float bv = minv[0][tid]; int bi = mini[0][tid];
    for (int w2 = 1; w2 < 8; ++w2)
      if (minv[w2][tid] < bv) { bv = minv[w2][tid]; bi = mini[w2][tid]; }
    fidx[tid] = bi;
    idxbuf[(base + tid) * QQ + q] = bi;
  }
  __syncthreads();
  float csum = 0.0f;
  for (int i = tid; i < 16 * 128; i += 256) {
    int r = i >> 7, c = i & 127;
    float code = cbf[((size_t)q * KK + fidx[r]) * 128 + c];
    float rr = arf[i];
    float dq = code - rr;
    csum += dq * dq;
    size_t go = (size_t)(base + r) * 128 + c;
    quant[go] += code;
    residual[go] = rr - code;
    tin[((size_t)(base + r) * QQ + q) * 128 + c] = code;
  }
  cred[tid] = csum; __syncthreads();
  for (int s = 128; s > 0; s >>= 1) { if (tid < s) cred[tid] += cred[tid + s]; __syncthreads(); }
  if (tid == 0) atomicAdd(&commit[q], cred[0]);
}

// token projection: tin[1536x128] @ out_W[128x512] + out_b
__global__ void k_tokgemm(const float* __restrict__ tin, const half_t* __restrict__ WoutPk,
                          const float* __restrict__ ob, float* __restrict__ pre) {
  __shared__ __align__(32) half_t ta[128 * 128];
  int base = blockIdx.x * 128;
  int tid = threadIdx.x;
  for (int i = tid; i < 128 * 32; i += 256) {
    float4 v = ((const float4*)tin)[(size_t)base * 32 + i];
    v4h h = {(half_t)v.x, (half_t)v.y, (half_t)v.z, (half_t)v.w};
    *(v4h*)(ta + i * 4) = h;
  }
  __syncthreads();
  int w = tid >> 5, lane = tid & 31, rb = w << 4, ro = (lane >> 4) << 3;
  v16h a4[4];
#pragma unroll
  for (int kk = 0; kk < 4; ++kk) a4[kk] = load_a_lds(ta, 128, rb, kk * 32);
  auto store = [&](v8f acc, int ct) {
    int colb = ct * 16 + (lane & 15);
    float bv = ob[colb];
#pragma unroll
    for (int i = 0; i < 8; ++i)
      pre[(size_t)(base + rb + i + ro) * FDD + colb] = acc[i] + bv;
  };
  v16h bA[4], bB[4];
#pragma unroll
  for (int kk = 0; kk < 4; ++kk) bA[kk] = load_b_pk(WoutPk, 32, kk, 0);
  for (int cp = 0; cp < 16; ++cp) {
    int ct0 = cp * 2;
#pragma unroll
    for (int kk = 0; kk < 4; ++kk) bB[kk] = load_b_pk(WoutPk, 32, kk, ct0 + 1);
    v8f acc = {};
#pragma unroll
    for (int kk = 0; kk < 4; ++kk) acc = wmma_f16(a4[kk], bA[kk], acc);
    store(acc, ct0);
    if (cp < 15) {
#pragma unroll
      for (int kk = 0; kk < 4; ++kk) bA[kk] = load_b_pk(WoutPk, 32, kk, ct0 + 2);
    }
    v8f acc2 = {};
#pragma unroll
    for (int kk = 0; kk < 4; ++kk) acc2 = wmma_f16(a4[kk], bB[kk], acc2);
    store(acc2, ct0 + 1);
  }
}

// layernorm over FD=512, write to output
__global__ void k_tokln(const float* __restrict__ pre, const float* __restrict__ og,
                        const float* __restrict__ ob, float* __restrict__ out) {
  __shared__ float red[256];
  int row = blockIdx.x, t = threadIdx.x;
  float x1 = pre[(size_t)row * FDD + t];
  float x2 = pre[(size_t)row * FDD + t + 256];
  red[t] = x1 + x2; __syncthreads();
  for (int s = 128; s > 0; s >>= 1) { if (t < s) red[t] += red[t + s]; __syncthreads(); }
  float m = red[0] * (1.0f / FDD); __syncthreads();
  float d1 = x1 - m, d2 = x2 - m;
  red[t] = d1 * d1 + d2 * d2; __syncthreads();
  for (int s = 128; s > 0; s >>= 1) { if (t < s) red[t] += red[t + s]; __syncthreads(); }
  float rs = rsqrtf(red[0] * (1.0f / FDD) + LN_EPS);
  out[(size_t)row * FDD + t]       = d1 * rs * og[t] + ob[t];
  out[(size_t)row * FDD + t + 256] = d2 * rs * og[t + 256] + ob[t + 256];
}

// losses + index copy
__global__ void k_final(const float* __restrict__ gn, const float* __restrict__ quant,
                        const float* __restrict__ commit, const int* __restrict__ idxbuf,
                        float* __restrict__ out) {
  __shared__ float red[256];
  int t = threadIdx.x;
  float rs = 0.0f;
  for (int i = t; i < GG * DD; i += 256) {
    float d = gn[i] - quant[i];
    rs += d * d;
  }
  red[t] = rs; __syncthreads();
  for (int s = 128; s > 0; s >>= 1) { if (t < s) red[t] += red[t + s]; __syncthreads(); }
  if (t == 0) {
    float cm = (commit[0] + commit[1] + commit[2]) * (1.0f / (3.0f * GG * DD));
    float rl = red[0] * (1.0f / (GG * DD));
    out[(size_t)GG * QQ * FDD + GG * QQ] = cm + rl;
  }
  int* ip = (int*)(out + (size_t)GG * QQ * FDD);
  for (int i = t; i < GG * QQ; i += 256) ip[i] = idxbuf[i];
}

extern "C" void kernel_launch(void* const* d_in, const int* in_sizes, int n_in,
                              void* d_out, int out_size, void* d_ws, size_t ws_size,
                              hipStream_t stream) {
  (void)in_sizes; (void)n_in; (void)out_size; (void)ws_size;
  const float* x     = (const float*)d_in[0];
  const int*   ei    = (const int*)d_in[1];
  const float* ea    = (const float*)d_in[2];
  const int*   batch = (const int*)d_in[3];
  const float* Wl    = (const float*)d_in[4];
  const float* bl    = (const float*)d_in[5];
  const float* Wr    = (const float*)d_in[6];
  const float* br    = (const float*)d_in[7];
  const float* We    = (const float*)d_in[8];
  const float* att   = (const float*)d_in[9];
  const float* cbias = (const float*)d_in[10];
  const float* Wad   = (const float*)d_in[11];
  const float* ab    = (const float*)d_in[12];
  const float* ng    = (const float*)d_in[13];
  const float* nb    = (const float*)d_in[14];
  const float* cbf   = (const float*)d_in[15];
  const float* Wout  = (const float*)d_in[16];
  const float* ob    = (const float*)d_in[17];
  const float* ogn   = (const float*)d_in[18];
  const float* obn   = (const float*)d_in[19];
  float* out = (float*)d_out;

  char* ws = (char*)d_ws;
  size_t o = 0;
  auto alloc = [&](size_t bytes) -> char* {
    char* p = ws + o;
    o = (o + bytes + 255) & ~(size_t)255;
    return p;
  };
  half_t* WlPk   = (half_t*)alloc(DD * DD * 2);
  half_t* WrPk   = (half_t*)alloc(DD * DD * 2);
  half_t* WePk   = (half_t*)alloc(DD * DD * 2);
  half_t* WadPk  = (half_t*)alloc(DD * DD * 2);
  half_t* WoutPk = (half_t*)alloc(DD * FDD * 2);
  half_t* cbPk   = (half_t*)alloc((size_t)QQ * DD * KK * 2);
  float*  cbsq   = (float*)alloc(QQ * KK * 4);
  float*  colsum = (float*)alloc(DD * 4);
  float*  me     = (float*)alloc(DD * 4);
  float*  xl     = (float*)alloc((size_t)NNP * DD * 4);
  float*  xr     = (float*)alloc((size_t)NNP * DD * 4);
  float*  scores = (float*)alloc((size_t)EEP * 4);
  float*  exv    = (float*)alloc((size_t)EE * 4);
  float*  ssel   = (float*)alloc((size_t)NN * 4);
  float*  exsel  = (float*)alloc((size_t)NN * 4);
  float*  smax   = (float*)alloc((size_t)(NN + 1) * 4);
  float*  denom  = (float*)alloc((size_t)(NN + 1) * 4);
  float*  outh   = (float*)alloc((size_t)NN * DD * 4);
  float*  gsum   = (float*)alloc(GG * DD * 4);
  float*  gcnt   = (float*)alloc(GG * 4);
  float*  gn     = (float*)alloc(GG * DD * 4);
  float*  resid  = (float*)alloc(GG * DD * 4);
  float*  quant  = (float*)alloc(GG * DD * 4);
  float*  tin    = (float*)alloc((size_t)GG * QQ * DD * 4);
  int*    idxbuf = (int*)alloc(GG * QQ * 4);
  float*  pre    = (float*)alloc((size_t)GG * QQ * FDD * 4);
  float*  lacc   = (float*)alloc(16);

  k_init<<<(NN * DD + 255) / 256, 256, 0, stream>>>(smax, denom, outh, gsum, gcnt, colsum, lacc);
  k_prep<<<768, 256, 0, stream>>>(Wl, Wr, We, Wad, Wout, cbf, WlPk, WrPk, WePk, WadPk, WoutPk, cbPk, cbsq);
  k_colsum<<<512, 256, 0, stream>>>(ea, colsum);
  k_meanemb<<<1, 128, 0, stream>>>(colsum, We, me);
  k_nodexf<<<NNP / 128, 256, 0, stream>>>(x, WlPk, WrPk, bl, br, xl, xr);
  k_edgescore<<<EEP / 128, 256, 0, stream>>>(ea, ei, WePk, xl, xr, att, scores, smax);
  k_selfscore<<<(NN * 32 + 255) / 256, 256, 0, stream>>>(xl, xr, me, att, ssel, smax);
  k_pass2<<<(EE + 255) / 256, 256, 0, stream>>>(scores, ssel, ei, smax, exv, exsel, denom);
  k_pass3e<<<(int)(((long long)EE * 32 + 255) / 256), 256, 0, stream>>>(ei, exv, denom, xl, outh);
  k_pass3s<<<(NN * 32 + 255) / 256, 256, 0, stream>>>(exsel, denom, xl, outh);
  k_pool<<<(NN * 32 + 255) / 256, 256, 0, stream>>>(outh, cbias, batch, gsum, gcnt);
  k_adapter<<<GG / 128, 256, 0, stream>>>(gsum, gcnt, WadPk, ab, gn);
  k_adln<<<GG, 128, 0, stream>>>(gn, ng, nb, resid, quant);
  for (int q = 0; q < QQ; ++q)
    k_vq<<<GG / 16, 256, 0, stream>>>(q, resid, cbPk, cbsq, cbf, quant, tin, idxbuf, lacc);
  k_tokgemm<<<(GG * QQ) / 128, 256, 0, stream>>>(tin, WoutPk, ob, pre);
  k_tokln<<<GG * QQ, 256, 0, stream>>>(pre, ogn, obn, out);
  k_final<<<1, 256, 0, stream>>>(gn, quant, lacc, idxbuf, out);
}
